// GatedGCNConv_65867618451767
// MI455X (gfx1250) — matrix operand order repaired and checked by reading, not compiled, not run on hardware
//
#include <hip/hip_runtime.h>

constexpr int DIM   = 64;
constexpr int NATOM = 250000;
constexpr int NBOND = 250000;
constexpr int NMOL  = 10000;
constexpr float BNEPS = 1e-5f;

typedef __attribute__((ext_vector_type(16))) __bf16 v16bf;
typedef __attribute__((ext_vector_type(8)))  float  v8f;

// ---------------------------------------------------------------------------
// WMMA fragment layouts (bf16 16x16x32, wave32).
// A (16x32, MxK): lanes 0-15 = rows, elems 0..7 -> K = half*8 + 0..7,
//                 elems 8..15 -> K = 16 + half*8 + 0..7   (half = lane>>4)
// B (32x16, KxN): lane&15 = col, elems i -> same K as A (half*8 + i, or 16 + half*8 + i-8)
// C/D (16x16 f32): elem j, lane l -> row = j + 8*(l>>4), col = l&15
//
// Weights are pre-packed to bf16 in fragment order:
//   Wp[((wi*4 + ct)*32 + lane)*32 + ks*16 + i]
// so each lane's B fragment for one (wi,ct) pair is 64 contiguous bytes.
// ---------------------------------------------------------------------------
__device__ __forceinline__ v16bf load_a_frag(const float* __restrict__ X,
                                             int row0, int ks, int lane) {
    const int m = lane & 15, half = lane >> 4;
    const float* p = X + (size_t)(row0 + m) * DIM + ks * 32 + half * 8;
    v16bf a;
#pragma unroll
    for (int i = 0; i < 8; ++i) a[i] = (__bf16)p[i];
#pragma unroll
    for (int i = 0; i < 8; ++i) a[8 + i] = (__bf16)p[16 + i];
    return a;
}

__device__ __forceinline__ v8f tile_gemm(v16bf a0, v16bf a1,
                                         const __bf16* __restrict__ Wp,
                                         int wi, int ct, int lane) {
    const __bf16* p = Wp + (((size_t)(wi * 4 + ct) * 32) + lane) * 32;
    const v16bf b0 = *(const v16bf*)p;
    const v16bf b1 = *(const v16bf*)(p + 16);
    v8f acc = {};
    acc = __builtin_amdgcn_wmma_f32_16x16x32_bf16(false, a0, false, b0,
                                                  (short)0, acc, false, false);
    acc = __builtin_amdgcn_wmma_f32_16x16x32_bf16(false, a1, false, b1,
                                                  (short)0, acc, false, false);
    return acc;
}

// Convert all 9 weight matrices into packed bf16 B-fragments.
__global__ void pack_weights_kernel(const float* __restrict__ Ws,
                                    __bf16* __restrict__ Wp) {
    const int idx = blockIdx.x * blockDim.x + threadIdx.x;   // 9*4*1024
    if (idx >= 9 * 4 * 1024) return;
    const int wi = idx >> 12;
    const int r  = idx & 4095;
    const int ct = r >> 10;
    const int r2 = r & 1023;
    const int lane = r2 >> 5;
    const int r3 = r2 & 31;
    const int ks = r3 >> 4;
    const int i  = r3 & 15;
    const int n = lane & 15, khalf = lane >> 4;
    const float v = Ws[(size_t)wi * DIM * DIM +
                       (size_t)(ks * 32 + (i < 8 ? khalf * 8 + i : 16 + khalf * 8 + (i - 8))) * DIM + ct * 16 + n];
    Wp[idx] = (__bf16)v;
}

// ---------------------------------------------------------------------------
// Three linears sharing one input read: O_k = X @ W[i_k] + b[i_k]
// ---------------------------------------------------------------------------
__global__ void gemm3_kernel(const float* __restrict__ X,
                             const __bf16* __restrict__ Wp,
                             const float* __restrict__ bs,
                             int i0, int i1, int i2,
                             float* __restrict__ O0, float* __restrict__ O1,
                             float* __restrict__ O2, int ntiles) {
    const int lane = threadIdx.x & 31;
    const int tile = blockIdx.x * (blockDim.x >> 5) + (threadIdx.x >> 5);
    if (tile >= ntiles) return;
    const int row0 = tile << 4;
    const v16bf a0 = load_a_frag(X, row0, 0, lane);
    const v16bf a1 = load_a_frag(X, row0, 1, lane);
    const int half = lane >> 4, n = lane & 15;
    const int widx[3] = {i0, i1, i2};
    float* const outs[3] = {O0, O1, O2};
    for (int w = 0; w < 3; ++w) {
        const float* bv = bs + widx[w] * DIM;
        float* out = outs[w];
#pragma unroll
        for (int ct = 0; ct < 4; ++ct) {
            v8f acc = tile_gemm(a0, a1, Wp, widx[w], ct, lane);
            const int col = ct * 16 + n;
            const float bias = bv[col];
#pragma unroll
            for (int j = 0; j < 8; ++j)
                out[(size_t)(row0 + half * 8 + j) * DIM + col] = acc[j] + bias;
        }
    }
}

// ---------------------------------------------------------------------------
// Bond update GEMM:  e_pre = (e@W1 + b1 + Ah[bu] + Ah[bv] + Cu[b2m]) * norm_bond
// ---------------------------------------------------------------------------
__global__ void gemm_e_bond_kernel(const float* __restrict__ E,
                                   const __bf16* __restrict__ Wp,
                                   const float* __restrict__ bs,
                                   const float* __restrict__ Ah,
                                   const float* __restrict__ Cu,
                                   const int* __restrict__ bond_u,
                                   const int* __restrict__ bond_v,
                                   const int* __restrict__ bond2mol,
                                   const float* __restrict__ norm_bond,
                                   float* __restrict__ e_pre, int ntiles) {
    const int lane = threadIdx.x & 31;
    const int tile = blockIdx.x * (blockDim.x >> 5) + (threadIdx.x >> 5);
    if (tile >= ntiles) return;
    const int row0 = tile << 4;
    const v16bf a0 = load_a_frag(E, row0, 0, lane);
    const v16bf a1 = load_a_frag(E, row0, 1, lane);
    const int half = lane >> 4, n = lane & 15;
    const float* bv = bs + 1 * DIM;
    int bu[8], bw[8], bm[8];
    float nb[8];
#pragma unroll
    for (int j = 0; j < 8; ++j) {
        const int r = row0 + half * 8 + j;
        bu[j] = bond_u[r]; bw[j] = bond_v[r]; bm[j] = bond2mol[r];
        nb[j] = norm_bond[r];
    }
#pragma unroll
    for (int ct = 0; ct < 4; ++ct) {
        v8f acc = tile_gemm(a0, a1, Wp, 1, ct, lane);
        const int col = ct * 16 + n;
        const float bias = bv[col];
#pragma unroll
        for (int j = 0; j < 8; ++j) {
            const int r = row0 + half * 8 + j;
            float v = acc[j] + bias
                    + Ah[(size_t)bu[j] * DIM + col]
                    + Ah[(size_t)bw[j] * DIM + col]
                    + Cu[(size_t)bm[j] * DIM + col];
            e_pre[(size_t)r * DIM + col] = v * nb[j];
        }
    }
}

// ---------------------------------------------------------------------------
// Global-update GEMM with fused segment-sum scatter:
//   acc_out[idx[r]] += (X@W[wi] + b[wi])[r]
// ---------------------------------------------------------------------------
__global__ void gemm_scatter_kernel(const float* __restrict__ X,
                                    const __bf16* __restrict__ Wp,
                                    const float* __restrict__ bs, int wi,
                                    const int* __restrict__ idx,
                                    float* __restrict__ acc_out, int ntiles) {
    const int lane = threadIdx.x & 31;
    const int tile = blockIdx.x * (blockDim.x >> 5) + (threadIdx.x >> 5);
    if (tile >= ntiles) return;
    const int row0 = tile << 4;
    const v16bf a0 = load_a_frag(X, row0, 0, lane);
    const v16bf a1 = load_a_frag(X, row0, 1, lane);
    const int half = lane >> 4, n = lane & 15;
    const float* bv = bs + wi * DIM;
    int mol[8];
#pragma unroll
    for (int j = 0; j < 8; ++j) mol[j] = idx[row0 + half * 8 + j];
#pragma unroll
    for (int ct = 0; ct < 4; ++ct) {
        v8f acc = tile_gemm(a0, a1, Wp, wi, ct, lane);
        const int col = ct * 16 + n;
        const float bias = bv[col];
#pragma unroll
        for (int j = 0; j < 8; ++j)
            atomicAdd(&acc_out[(size_t)mol[j] * DIM + col], acc[j] + bias);
    }
}

// ---------------------------------------------------------------------------
// Elementwise / reduction helpers
// ---------------------------------------------------------------------------
__global__ void zero_kernel(float* __restrict__ p, size_t n) {
    size_t i = blockIdx.x * (size_t)blockDim.x + threadIdx.x;
    const size_t step = (size_t)gridDim.x * blockDim.x;
    for (; i < n; i += step) p[i] = 0.f;
}

__global__ void colstats_kernel(const float* __restrict__ x, int n,
                                float* __restrict__ stats) {
    const int c = threadIdx.x & 63;
    const int rpb = blockDim.x >> 6;
    int r = blockIdx.x * rpb + (threadIdx.x >> 6);
    const int rstep = gridDim.x * rpb;
    float s = 0.f, s2 = 0.f;
    for (; r < n; r += rstep) {
        const float v = x[(size_t)r * DIM + c];
        s += v; s2 += v * v;
    }
    atomicAdd(&stats[c], s);
    atomicAdd(&stats[DIM + c], s2);
}

__global__ void bn_elu_kernel(float* __restrict__ x, int n,
                              const float* __restrict__ stats,
                              const float* __restrict__ gamma,
                              const float* __restrict__ beta) {
    const size_t i = blockIdx.x * (size_t)blockDim.x + threadIdx.x;
    if (i >= (size_t)n * DIM) return;
    const int c = (int)(i & 63);
    const float inv_n = 1.0f / (float)n;
    const float mean = stats[c] * inv_n;
    const float var = stats[DIM + c] * inv_n - mean * mean;
    float y = (x[i] - mean) * rsqrtf(var + BNEPS) * gamma[c] + beta[c];
    x[i] = y > 0.f ? y : (expf(y) - 1.f);
}

__global__ void scatter_atom_kernel(const float* __restrict__ e_new,
                                    const float* __restrict__ Eh,
                                    const int* __restrict__ bond_u,
                                    const int* __restrict__ bond_v,
                                    float* __restrict__ num,
                                    float* __restrict__ den) {
    const size_t i = blockIdx.x * (size_t)blockDim.x + threadIdx.x;
    if (i >= (size_t)NBOND * DIM) return;
    const int b = (int)(i >> 6), c = (int)(i & 63);
    const float s = 1.f / (1.f + expf(-e_new[i]));
    const int bu = bond_u[b], bv = bond_v[b];
    atomicAdd(&num[(size_t)bu * DIM + c], s * Eh[(size_t)bv * DIM + c]);
    atomicAdd(&den[(size_t)bu * DIM + c], s);
    atomicAdd(&num[(size_t)bv * DIM + c], s * Eh[(size_t)bu * DIM + c]);
    atomicAdd(&den[(size_t)bv * DIM + c], s);
}

__global__ void atom_combine_kernel(const float* __restrict__ Dh,
                                    const float* __restrict__ num,
                                    const float* __restrict__ den,
                                    const float* __restrict__ Fu,
                                    const int* __restrict__ atom2mol,
                                    const float* __restrict__ norm_atom,
                                    float* __restrict__ h_pre) {
    const size_t i = blockIdx.x * (size_t)blockDim.x + threadIdx.x;
    if (i >= (size_t)NATOM * DIM) return;
    const int a = (int)(i >> 6), c = (int)(i & 63);
    float v = Dh[i] + num[i] / (den[i] + 1e-6f)
            + Fu[(size_t)atom2mol[a] * DIM + c];
    h_pre[i] = v * norm_atom[a];
}

__global__ void count_kernel(const int* __restrict__ idx, int n,
                             float* __restrict__ cnt) {
    const int i = blockIdx.x * blockDim.x + threadIdx.x;
    if (i < n) atomicAdd(&cnt[idx[i]], 1.0f);
}

__global__ void u_final_kernel(const float* __restrict__ ua,
                               const float* __restrict__ ub,
                               const float* __restrict__ cnta,
                               const float* __restrict__ cntb,
                               const float* __restrict__ Iu,
                               float* __restrict__ u_pre) {
    const size_t i = blockIdx.x * (size_t)blockDim.x + threadIdx.x;
    if (i >= (size_t)NMOL * DIM) return;
    const int m = (int)(i >> 6);
    const float ca = fmaxf(cnta[m], 1.f), cb = fmaxf(cntb[m], 1.f);
    u_pre[i] = ua[i] / ca + ub[i] / cb + Iu[i];
}

// ---------------------------------------------------------------------------
extern "C" void kernel_launch(void* const* d_in, const int* in_sizes, int n_in,
                              void* d_out, int out_size, void* d_ws, size_t ws_size,
                              hipStream_t stream) {
    const float* h         = (const float*)d_in[0];
    const float* e         = (const float*)d_in[1];
    const float* u         = (const float*)d_in[2];
    const float* norm_atom = (const float*)d_in[3];
    const float* norm_bond = (const float*)d_in[4];
    const float* Ws        = (const float*)d_in[5];
    const float* bs        = (const float*)d_in[6];
    const float* bn_gamma  = (const float*)d_in[7];
    const float* bn_beta   = (const float*)d_in[8];
    const int* bond_u      = (const int*)d_in[9];
    const int* bond_v      = (const int*)d_in[10];
    const int* atom2mol    = (const int*)d_in[11];
    const int* bond2mol    = (const int*)d_in[12];

    float* out   = (float*)d_out;
    float* h_new = out;                                   // NATOM*DIM
    float* e_new = out + (size_t)NATOM * DIM;             // NBOND*DIM
    float* u_new = e_new + (size_t)NBOND * DIM;           // NMOL*DIM

    float* ws = (float*)d_ws;
    size_t off = 0;
    float* num     = ws + off; off += (size_t)NATOM * DIM;
    float* den     = ws + off; off += (size_t)NATOM * DIM;
    float* uacc_a  = ws + off; off += (size_t)NMOL * DIM;
    float* uacc_b  = ws + off; off += (size_t)NMOL * DIM;
    float* cnt_a   = ws + off; off += NMOL;
    float* cnt_b   = ws + off; off += NMOL;
    float* stats_e = ws + off; off += 2 * DIM;
    float* stats_h = ws + off; off += 2 * DIM;
    float* stats_u = ws + off; off += 2 * DIM;
    const size_t zcount = off;                            // zeroed each call
    float* Ah = ws + off; off += (size_t)NATOM * DIM;
    float* Dh = ws + off; off += (size_t)NATOM * DIM;
    float* Eh = ws + off; off += (size_t)NATOM * DIM;
    float* Cu = ws + off; off += (size_t)NMOL * DIM;
    float* Fu = ws + off; off += (size_t)NMOL * DIM;
    float* Iu = ws + off; off += (size_t)NMOL * DIM;
    __bf16* Wp = (__bf16*)(ws + off);                     // 9*4*1024 bf16

    const int atiles = NATOM / 16, btiles = NBOND / 16, mtiles = NMOL / 16;
    const int EW = 256;

    zero_kernel<<<4096, EW, 0, stream>>>(ws, zcount);
    pack_weights_kernel<<<(9 * 4 * 1024 + EW - 1) / EW, EW, 0, stream>>>(Ws, Wp);

    // u-side linears: Cu, Fu, Iu
    gemm3_kernel<<<(mtiles + 3) / 4, 128, 0, stream>>>(u, Wp, bs, 2, 5, 8,
                                                       Cu, Fu, Iu, mtiles);
    // h-side linears: Ah, Dh, Eh
    gemm3_kernel<<<(atiles + 3) / 4, 128, 0, stream>>>(h, Wp, bs, 0, 3, 4,
                                                       Ah, Dh, Eh, atiles);
    // bond update (fused gathers + graph norm), pre-BN into d_out
    gemm_e_bond_kernel<<<(btiles + 3) / 4, 128, 0, stream>>>(
        e, Wp, bs, Ah, Cu, bond_u, bond_v, bond2mol, norm_bond, e_new, btiles);
    colstats_kernel<<<1024, EW, 0, stream>>>(e_new, NBOND, stats_e);
    bn_elu_kernel<<<(NBOND * DIM + EW - 1) / EW, EW, 0, stream>>>(
        e_new, NBOND, stats_e, bn_gamma + 0 * DIM, bn_beta + 0 * DIM);

    // gated aggregation to atoms
    scatter_atom_kernel<<<(NBOND * DIM + EW - 1) / EW, EW, 0, stream>>>(
        e_new, Eh, bond_u, bond_v, num, den);
    atom_combine_kernel<<<(NATOM * DIM + EW - 1) / EW, EW, 0, stream>>>(
        Dh, num, den, Fu, atom2mol, norm_atom, h_new);
    colstats_kernel<<<1024, EW, 0, stream>>>(h_new, NATOM, stats_h);
    bn_elu_kernel<<<(NATOM * DIM + EW - 1) / EW, EW, 0, stream>>>(
        h_new, NATOM, stats_h, bn_gamma + 1 * DIM, bn_beta + 1 * DIM);

    // global update: per-mol means of Gh / He + Iu
    count_kernel<<<(NATOM + EW - 1) / EW, EW, 0, stream>>>(atom2mol, NATOM, cnt_a);
    count_kernel<<<(NBOND + EW - 1) / EW, EW, 0, stream>>>(bond2mol, NBOND, cnt_b);
    gemm_scatter_kernel<<<(atiles + 3) / 4, 128, 0, stream>>>(
        h_new, Wp, bs, 6, atom2mol, uacc_a, atiles);
    gemm_scatter_kernel<<<(btiles + 3) / 4, 128, 0, stream>>>(
        e_new, Wp, bs, 7, bond2mol, uacc_b, btiles);
    u_final_kernel<<<(NMOL * DIM + EW - 1) / EW, EW, 0, stream>>>(
        uacc_a, uacc_b, cnt_a, cnt_b, Iu, u_new);
    colstats_kernel<<<256, EW, 0, stream>>>(u_new, NMOL, stats_u);
    bn_elu_kernel<<<(NMOL * DIM + EW - 1) / EW, EW, 0, stream>>>(
        u_new, NMOL, stats_u, bn_gamma + 2 * DIM, bn_beta + 2 * DIM);
}
